// Model_2997887173074
// MI455X (gfx1250) — compile-verified, matrix operand order checked by reading, not run on hardware
//
#include <hip/hip_runtime.h>
#include <hip/hip_bf16.h>
#include <math.h>

// ---------------------------------------------------------------------------
// Types / WMMA helpers (CDNA5 gfx1250, wave32)
// ---------------------------------------------------------------------------
typedef _Float16 h16;
typedef __attribute__((ext_vector_type(8)))  _Float16 v8h;
typedef __attribute__((ext_vector_type(16))) _Float16 v16h;
typedef __attribute__((ext_vector_type(8)))  float    v8f;

#define DEV __device__ __forceinline__

static constexpr int Bz    = 32;
static constexpr int Tt    = 512;
static constexpr int Nn    = 64;     // nodes
static constexpr int WINL  = 30;
static constexpr int STEPW = 2;
static constexpr int TOPKk = 20;
static constexpr int Sw    = 242;    // (512-30)/2 + 1
static constexpr int Dm    = 128;
static constexpr int HEADS = 8;
static constexpr int FFd   = 256;
static constexpr int Mrows = Bz * Sw; // 7744 = 242*32

DEV v8f vzero8() { v8f z; for (int i = 0; i < 8; ++i) z[i] = 0.f; return z; }

// Load a 16x32 f16 fragment whose "row" dimension is contiguous in K.
// Works for A-operand (row-major MxK, row = m) and for B-operand when the
// matrix is stored as NxK row-major (row = n), since B[k][n] column n is then
// contiguous in k. ISA layout: lanes 0-15 hold K {k0..k0+7, k0+16..k0+23},
// lanes 16-31 hold K {k0+8..k0+15, k0+24..k0+31}.
DEV v16h frag_ld(const h16* base, int ld, int row0, int k0) {
  int l = threadIdx.x & 31;
  const h16* p = base + (size_t)(row0 + (l & 15)) * ld + k0 + ((l >> 4) << 3);
  v8h lo = *(const v8h*)p;
  v8h hi = *(const v8h*)(p + 16);
  return __builtin_shufflevector(lo, hi, 0,1,2,3,4,5,6,7,8,9,10,11,12,13,14,15);
}

DEV v8f wmma_f16(v16h a, v16h b, v8f c) {
  return __builtin_amdgcn_wmma_f32_16x16x32_f16(false, a, false, b, (short)0, c,
                                                false, false);
}

// ---------------------------------------------------------------------------
// Weight conversion: f32 (N x Ksrc) -> f16 (N x K) row-major, zero-padded in K
// ---------------------------------------------------------------------------
__global__ void cvt_w(const float* __restrict__ src, h16* __restrict__ dst,
                      int N, int K, int Ksrc) {
  int i = blockIdx.x * 256 + threadIdx.x;
  if (i >= N * K) return;
  int n = i / K, k = i - n * K;
  dst[i] = (h16)((k < Ksrc) ? src[n * Ksrc + k] : 0.f);
}

// ---------------------------------------------------------------------------
// Fused per-window graph kernel: corr -> topk adjacency -> GIN x2 -> readout
// One block per (b,s) window.  256 threads = 8 waves.  Dynamic LDS.
// ---------------------------------------------------------------------------
// LDS layout (bytes)
static constexpr unsigned O_XM   = 0;            // 1920 f32
static constexpr unsigned O_MEAN = 7680;         // 64 f32
static constexpr unsigned O_STDC = 7936;         // 64 f32 (corr std)
static constexpr unsigned O_STDF = 8192;         // 64 f32 (feature std)
static constexpr unsigned O_DEG  = 8448;         // 64 f32
static constexpr unsigned O_SC   = 8704;         // 64 f32 (readout scores)
static constexpr unsigned O_MASK = 8960;         // 128 u32
static constexpr unsigned O_CORR = 9472;         // 4096 f32
static constexpr unsigned O_AH   = 25856;        // 4096 h16
static constexpr unsigned O_HF   = 34048;        // 8192 f32
static constexpr unsigned O_HH   = 66816;        // 8192 h16
static constexpr unsigned O_HT   = 83200;        // 8192 h16 (H transposed)
static constexpr unsigned O_HPRE = 99584;        // 8192 h16
static constexpr unsigned O_TH   = 115968;       // 8192 h16
static constexpr unsigned O_SX   = 132352;       // 8192 f32 (also A f32)
static constexpr unsigned GRAPH_LDS = 165120;

DEV void gin_layer(const h16* Ah, float* Hf, h16* Hh, h16* Ht, h16* hpre,
                   h16* th, const h16* W1h, const h16* W2h,
                   const float* __restrict__ b1, const float* __restrict__ b2,
                   float eps) {
  int wv = threadIdx.x >> 5;
  int l  = threadIdx.x & 31;
  int ln = l & 15, lm = (l >> 4) << 3;
  // step 1: agg = A(64x64) @ H(64x128)   (B-operand via Ht rows)
  for (int j = 0; j < 4; ++j) {
    int idx = wv * 4 + j, Mt = idx >> 3, Nt = idx & 7;
    v8f c = vzero8();
    for (int k = 0; k < 64; k += 32)
      c = wmma_f16(frag_ld(Ah, 64, Mt * 16, k), frag_ld(Ht, 64, Nt * 16, k), c);
    int n = Nt * 16 + ln, mb = Mt * 16 + lm;
    #pragma unroll
    for (int r = 0; r < 8; ++r) {
      float v = (1.f + eps) * Hf[(mb + r) * Dm + n] + c[r];
      hpre[(mb + r) * Dm + n] = (h16)v;
    }
  }
  __syncthreads();
  // step 2: t = relu(hpre @ W1^T + b1), K = 128 (layer0 W1 zero-padded)
  for (int j = 0; j < 4; ++j) {
    int idx = wv * 4 + j, Mt = idx >> 3, Nt = idx & 7;
    v8f c = vzero8();
    for (int k = 0; k < Dm; k += 32)
      c = wmma_f16(frag_ld(hpre, Dm, Mt * 16, k), frag_ld(W1h, Dm, Nt * 16, k), c);
    int n = Nt * 16 + ln, mb = Mt * 16 + lm;
    #pragma unroll
    for (int r = 0; r < 8; ++r) {
      float v = fmaxf(c[r] + b1[n], 0.f);
      th[(mb + r) * Dm + n] = (h16)v;
    }
  }
  __syncthreads();
  // step 3: Hnew = t @ W2^T + b2  -> Hf (f32), Hh (f16), Ht (f16, transposed)
  for (int j = 0; j < 4; ++j) {
    int idx = wv * 4 + j, Mt = idx >> 3, Nt = idx & 7;
    v8f c = vzero8();
    for (int k = 0; k < Dm; k += 32)
      c = wmma_f16(frag_ld(th, Dm, Mt * 16, k), frag_ld(W2h, Dm, Nt * 16, k), c);
    int n = Nt * 16 + ln, mb = Mt * 16 + lm;
    #pragma unroll
    for (int r = 0; r < 8; ++r) {
      float v = c[r] + b2[n];
      int m = mb + r;
      Hf[m * Dm + n] = v;
      Hh[m * Dm + n] = (h16)v;
      Ht[n * 64 + m] = (h16)v;
    }
  }
  __syncthreads();
}

__global__ void graph_kernel(
    const float* __restrict__ x_enc, const float* __restrict__ nn_g,
    const float* __restrict__ nn_b, const h16* __restrict__ g0W1h,
    const h16* __restrict__ g0W2h, const float* __restrict__ g0b1,
    const float* __restrict__ g0b2, const float* __restrict__ g0eps,
    const h16* __restrict__ g1W1h, const h16* __restrict__ g1W2h,
    const float* __restrict__ g1b1, const float* __restrict__ g1b2,
    const float* __restrict__ g1eps, const h16* __restrict__ rWh,
    const float* __restrict__ rb, const float* __restrict__ ru,
    float* __restrict__ Gf, h16* __restrict__ Gh) {
  extern __shared__ char smem[];
  float*    xm    = (float*)(smem + O_XM);
  float*    cmean = (float*)(smem + O_MEAN);
  float*    cstdc = (float*)(smem + O_STDC);
  float*    cstdf = (float*)(smem + O_STDF);
  float*    degv  = (float*)(smem + O_DEG);
  float*    scv   = (float*)(smem + O_SC);
  unsigned* maskb = (unsigned*)(smem + O_MASK);
  float*    corr  = (float*)(smem + O_CORR);
  h16*      Ah    = (h16*)(smem + O_AH);
  float*    Hf    = (float*)(smem + O_HF);
  h16*      Hh    = (h16*)(smem + O_HH);
  h16*      Ht    = (h16*)(smem + O_HT);
  h16*      hpre  = (h16*)(smem + O_HPRE);
  h16*      th    = (h16*)(smem + O_TH);
  float*    Sx    = (float*)(smem + O_SX);   // also A(f32) early on
  float*    Af    = Sx;

  int w = blockIdx.x;
  int b = w / Sw, s = w - b * Sw;
  int start = s * STEPW;
  int t = threadIdx.x;

  // --- window load ---
  for (int i = t; i < WINL * Nn; i += 256) {
    int l = i >> 6, n = i & 63;
    xm[i] = x_enc[((size_t)b * Tt + start + l) * Nn + n];
  }
  __syncthreads();
  if (t < 64) {
    float m = 0.f;
    for (int l = 0; l < WINL; ++l) m += xm[l * 64 + t];
    cmean[t] = m / (float)WINL;
  }
  __syncthreads();
  for (int i = t; i < WINL * Nn; i += 256) xm[i] -= cmean[i & 63];
  __syncthreads();
  if (t < 64) {
    float ss = 0.f;
    for (int l = 0; l < WINL; ++l) { float d = xm[l * 64 + t]; ss += d * d; }
    float var_c = ss / (float)(WINL - 1) /* + ~1e-8 denom -> negligible */;
    float var_cc = ss / ((float)(WINL - 1) + 1e-8f);
    var_cc = fmaxf(var_cc, 1e-8f);
    cstdc[t] = sqrtf(var_cc);
    cstdf[t] = fmaxf(sqrtf(var_c), 1e-8f);
  }
  __syncthreads();

  // --- correlation ---
  for (int i = t; i < 64 * 64; i += 256) {
    int m = i >> 6, n = i & 63;
    float acc = 0.f;
    for (int l = 0; l < WINL; ++l) acc += xm[l * 64 + m] * xm[l * 64 + n];
    float c = acc / ((float)(WINL - 1) + 1e-8f);
    c = c / (cstdc[m] * cstdc[n]);
    if (!(fabsf(c) < 3.0e38f)) c = 0.f;   // nan / inf -> 0
    corr[i] = (m == n) ? 1.0f : c;
  }
  __syncthreads();

  // --- top-k per row (exact, first-index tie break) ---
  if (t < 64) {
    const float* row = corr + t * 64;
    unsigned m0 = 0u, m1 = 0u;
    for (int kk = 0; kk < TOPKk; ++kk) {
      float best = -1.f; int bi = 0;
      for (int j = 0; j < 64; ++j) {
        bool sel = (j < 32) ? ((m0 >> j) & 1u) : ((m1 >> (j - 32)) & 1u);
        if (sel) continue;
        float a = fabsf(row[j]);
        if (a > best) { best = a; bi = j; }
      }
      if (bi < 32) m0 |= (1u << bi); else m1 |= (1u << (bi - 32));
    }
    maskb[t * 2] = m0; maskb[t * 2 + 1] = m1;
  }
  __syncthreads();

  // --- adjacency A = max(corr*mask, (corr*mask)^T), diag=1 ---
  for (int i = t; i < 64 * 64; i += 256) {
    int m = i >> 6, n = i & 63;
    float c = corr[i];
    float pm = (float)((maskb[m * 2 + (n >> 5)] >> (n & 31)) & 1u);
    float pn = (float)((maskb[n * 2 + (m >> 5)] >> (m & 31)) & 1u);
    float a = fmaxf(c * pm, c * pn);
    if (m == n) a = 1.0f;
    Af[i] = a;
    Ah[i] = (h16)a;
  }
  __syncthreads();
  if (t < 64) {
    float d = 0.f;
    for (int j = 0; j < 64; ++j) d += Af[t * 64 + j];
    degv[t] = d - 1.0f;
  }
  // --- init H buffers ---
  for (int i = t; i < 64 * Dm; i += 256) { Hf[i] = 0.f; Hh[i] = (h16)0.f; Ht[i] = (h16)0.f; }
  __syncthreads();
  if (t < 64) {
    float f0 = cmean[t], f1 = cstdf[t], f2 = degv[t];
    float mu = (f0 + f1 + f2) * (1.f / 3.f);
    float d0 = f0 - mu, d1 = f1 - mu, d2 = f2 - mu;
    float va = (d0 * d0 + d1 * d1 + d2 * d2) * (1.f / 3.f);
    float inv = rsqrtf(va + 1e-5f);
    float h0 = d0 * inv * nn_g[0] + nn_b[0];
    float h1 = d1 * inv * nn_g[1] + nn_b[1];
    float h2 = d2 * inv * nn_g[2] + nn_b[2];
    Hf[t * Dm + 0] = h0; Hf[t * Dm + 1] = h1; Hf[t * Dm + 2] = h2;
    Hh[t * Dm + 0] = (h16)h0; Hh[t * Dm + 1] = (h16)h1; Hh[t * Dm + 2] = (h16)h2;
    Ht[0 * 64 + t] = (h16)h0; Ht[1 * 64 + t] = (h16)h1; Ht[2 * 64 + t] = (h16)h2;
  }
  __syncthreads();

  // --- GIN x2 ---
  gin_layer(Ah, Hf, Hh, Ht, hpre, th, g0W1h, g0W2h, g0b1, g0b2, g0eps[0]);
  gin_layer(Ah, Hf, Hh, Ht, hpre, th, g1W1h, g1W2h, g1b1, g1b2, g1eps[0]);

  // --- readout: Sx = tanh(H @ rW^T + rb) ---
  {
    int wv = threadIdx.x >> 5;
    int l  = threadIdx.x & 31;
    int ln = l & 15, lm = (l >> 4) << 3;
    for (int j = 0; j < 4; ++j) {
      int idx = wv * 4 + j, Mt = idx >> 3, Nt = idx & 7;
      v8f c = vzero8();
      for (int k = 0; k < Dm; k += 32)
        c = wmma_f16(frag_ld(Hh, Dm, Mt * 16, k), frag_ld(rWh, Dm, Nt * 16, k), c);
      int n = Nt * 16 + ln, mb = Mt * 16 + lm;
      #pragma unroll
      for (int r = 0; r < 8; ++r)
        Sx[(mb + r) * Dm + n] = tanhf(c[r] + rb[n]);
    }
  }
  __syncthreads();
  if (t < 64) {
    float a = 0.f;
    for (int d = 0; d < Dm; ++d) a += Sx[t * Dm + d] * ru[d];
    scv[t] = a;
  }
  __syncthreads();
  if (t < Dm) {
    float mx = -1e30f;
    for (int n = 0; n < 64; ++n) mx = fmaxf(mx, scv[n]);
    float den = 0.f;
    for (int n = 0; n < 64; ++n) den += expf(scv[n] - mx);
    float inv = 1.f / den;
    float g = 0.f;
    for (int n = 0; n < 64; ++n)
      g += expf(scv[n] - mx) * inv * Hf[n * Dm + t];
    // positional encoding
    float ex  = -(logf(10000.f) / (float)Dm) * (float)(t & ~1);
    float ang = (float)s * expf(ex);
    g += (t & 1) ? cosf(ang) : sinf(ang);
    Gf[(size_t)w * Dm + t] = g;
    Gh[(size_t)w * Dm + t] = (h16)g;
  }
}

// ---------------------------------------------------------------------------
// Generic WMMA GEMM: out = act(X(MxK,f16) @ W(NxK,f16)^T + bias)
// grid(M/32, N/64), block 256 (8 waves, 1 tile each). act: 0 none, 1 gelu, 2 tanh
// ---------------------------------------------------------------------------
__global__ void gemm_xwt(const h16* __restrict__ X, const h16* __restrict__ W,
                         const float* __restrict__ bias, float* __restrict__ outF,
                         h16* __restrict__ outH, int K, int N, int act) {
  int wv = threadIdx.x >> 5;
  int m0 = blockIdx.x * 32 + (wv >> 2) * 16;
  int n0 = blockIdx.y * 64 + (wv & 3) * 16;
  v8f c = vzero8();
  for (int k = 0; k < K; k += 32)
    c = wmma_f16(frag_ld(X, K, m0, k), frag_ld(W, K, n0, k), c);
  int l = threadIdx.x & 31;
  int n = n0 + (l & 15);
  int mb = m0 + ((l >> 4) << 3);
  float bv = bias ? bias[n] : 0.f;
  #pragma unroll
  for (int r = 0; r < 8; ++r) {
    float v = c[r] + bv;
    if (act == 1)      v = 0.5f * v * (1.f + erff(v * 0.70710678118f)); // exact gelu
    else if (act == 2) v = tanhf(v);
    size_t o = (size_t)(mb + r) * N + n;
    if (outF) outF[o] = v;
    if (outH) outH[o] = (h16)v;
  }
}

// ---------------------------------------------------------------------------
// Attention: one wave per (b, head, 16-query tile).  S padded 242->256,
// head dim padded 16->32 with zeros.  grid(16, HEADS, B), block 32.
// ---------------------------------------------------------------------------
static constexpr unsigned AO_Q  = 0;      // 16x32 h16   (1024 B)
static constexpr unsigned AO_K  = 1024;   // 256x32 h16  (16384 B)
static constexpr unsigned AO_VT = 17408;  // 16x256 h16  (8192 B)
static constexpr unsigned AO_P  = 25600;  // 16x256 h16  (8192 B)
static constexpr unsigned AO_S  = 33792;  // 16x256 f32  (16384 B)
static constexpr unsigned ATT_LDS = 50176;

__global__ void attn_kernel(const h16* __restrict__ QKV, h16* __restrict__ Oh) {
  extern __shared__ char smem[];
  h16*   Qm = (h16*)(smem + AO_Q);
  h16*   Km = (h16*)(smem + AO_K);
  h16*   Vt = (h16*)(smem + AO_VT);
  h16*   Pm = (h16*)(smem + AO_P);
  float* sc = (float*)(smem + AO_S);
  int qt = blockIdx.x, h = blockIdx.y, b = blockIdx.z;
  int lane = threadIdx.x & 31;

  for (int i = lane; i < 16 * 32; i += 32) {
    int r = i >> 5, c = i & 31;
    int ss = qt * 16 + r;
    h16 v = (h16)0.f;
    if (ss < Sw && c < 16) v = QKV[((size_t)(b * Sw + ss)) * 384 + h * 16 + c];
    Qm[i] = v;
  }
  for (int i = lane; i < 256 * 32; i += 32) {
    int r = i >> 5, c = i & 31;
    h16 v = (h16)0.f;
    if (r < Sw && c < 16) v = QKV[((size_t)(b * Sw + r)) * 384 + 128 + h * 16 + c];
    Km[i] = v;
  }
  for (int i = lane; i < 16 * 256; i += 32) {
    int d = i >> 8, r = i & 255;
    h16 v = (h16)0.f;
    if (r < Sw) v = QKV[((size_t)(b * Sw + r)) * 384 + 256 + h * 16 + d];
    Vt[i] = v;
  }
  __syncthreads();

  v16h qf = frag_ld(Qm, 32, 0, 0);
  for (int nt = 0; nt < 16; ++nt) {
    v8f c = wmma_f16(qf, frag_ld(Km, 32, nt * 16, 0), vzero8());
    int n = nt * 16 + (lane & 15), mb = (lane >> 4) << 3;
    #pragma unroll
    for (int r = 0; r < 8; ++r) sc[(mb + r) * 256 + n] = c[r] * 0.25f; // /sqrt(16)
  }
  __syncthreads();
  if (lane < 16) {
    float* row = sc + lane * 256;
    float mx = -1e30f;
    for (int j = 0; j < Sw; ++j) mx = fmaxf(mx, row[j]);
    float den = 0.f;
    for (int j = 0; j < Sw; ++j) { float e = expf(row[j] - mx); row[j] = e; den += e; }
    float inv = 1.f / den;
    for (int j = 0; j < 256; ++j)
      Pm[lane * 256 + j] = (h16)((j < Sw) ? row[j] * inv : 0.f);
  }
  __syncthreads();
  v8f c = vzero8();
  for (int k = 0; k < 256; k += 32)
    c = wmma_f16(frag_ld(Pm, 256, 0, k), frag_ld(Vt, 256, 0, k), c);
  int n = lane & 15, mb = (lane >> 4) << 3;
  #pragma unroll
  for (int r = 0; r < 8; ++r) {
    int ss = qt * 16 + mb + r;
    if (ss < Sw) Oh[((size_t)(b * Sw + ss)) * Dm + h * 16 + n] = (h16)c[r];
  }
}

// ---------------------------------------------------------------------------
// Residual + LayerNorm over 128 (in-place on Gf; also writes f16 copy)
// grid M, block 128
// ---------------------------------------------------------------------------
__global__ void resid_ln(const float* __restrict__ Gin, const float* __restrict__ add,
                         const float* __restrict__ g, const float* __restrict__ b,
                         float* __restrict__ Gf, h16* __restrict__ Gh) {
  __shared__ float buf[Dm];
  size_t row = blockIdx.x;
  int d = threadIdx.x;
  float v = Gin[row * Dm + d] + add[row * Dm + d];
  buf[d] = v;
  __syncthreads();
  float mu = 0.f;
  for (int i = 0; i < Dm; ++i) mu += buf[i];
  mu *= (1.f / Dm);
  float var = 0.f;
  for (int i = 0; i < Dm; ++i) { float x = buf[i] - mu; var += x * x; }
  var *= (1.f / Dm);
  float o = (v - mu) * rsqrtf(var + 1e-5f) * g[d] + b[d];
  Gf[row * Dm + d] = o;
  Gh[row * Dm + d] = (h16)o;
}

// ---------------------------------------------------------------------------
// Attention pooling + classifier.  grid B, block 256
// Tb = tanh(G @ pW^T + pb) precomputed (f32).
// ---------------------------------------------------------------------------
__global__ void pool_kernel(const float* __restrict__ Tb, const float* __restrict__ Gf,
                            const float* __restrict__ pv, const float* __restrict__ cW,
                            const float* __restrict__ cb, float* __restrict__ out) {
  __shared__ float scr[Sw];
  __shared__ float zz[Dm];
  int b = blockIdx.x, t = threadIdx.x;
  for (int s = t; s < Sw; s += 256) {
    const float* r = Tb + (size_t)(b * Sw + s) * Dm;
    float a = 0.f;
    for (int d = 0; d < Dm; ++d) a += r[d] * pv[d];
    scr[s] = a;
  }
  __syncthreads();
  if (t < Dm) {
    float mx = -1e30f;
    for (int s = 0; s < Sw; ++s) mx = fmaxf(mx, scr[s]);
    float den = 0.f;
    for (int s = 0; s < Sw; ++s) den += expf(scr[s] - mx);
    float inv = 1.f / den;
    float acc = 0.f;
    for (int s = 0; s < Sw; ++s)
      acc += expf(scr[s] - mx) * inv * Gf[(size_t)(b * Sw + s) * Dm + t];
    zz[t] = acc * cW[t];
  }
  __syncthreads();
  if (t == 0) {
    float o = 0.f;
    for (int d = 0; d < Dm; ++d) o += zz[d];
    o += cb[0];
    out[b] = 1.f / (1.f + expf(-o));
  }
}

// ---------------------------------------------------------------------------
// Host side
// ---------------------------------------------------------------------------
extern "C" void kernel_launch(void* const* d_in, const int* in_sizes, int n_in,
                              void* d_out, int out_size, void* d_ws, size_t ws_size,
                              hipStream_t stream) {
  (void)out_size; (void)ws_size;
  // locate x_enc (unique size) and params base (jax pytree flatten order)
  int base;
  const float* x_enc;
  if (in_sizes[0] == Bz * Tt * Nn) { x_enc = (const float*)d_in[0]; base = 1; }
  else { x_enc = (const float*)d_in[n_in - 1]; base = 0; }
  auto P = [&](int i) { return (const float*)d_in[base + i]; };
  // params flatten (dict keys sorted): cls, gin[2], layers[2], node_norm, pool, readout
  const float* cls_W = P(0);  const float* cls_b = P(1);
  const float* g0W1 = P(2);  const float* g0W2 = P(3);
  const float* g0b1 = P(4);  const float* g0b2 = P(5);  const float* g0eps = P(6);
  const float* g1W1 = P(7);  const float* g1W2 = P(8);
  const float* g1b1 = P(9);  const float* g1b2 = P(10); const float* g1eps = P(11);
  const float* Lp[2][12];   // W1,W2,Wo,Wqkv,b1,b2,bo,bqkv,ln1_b,ln1_g,ln2_b,ln2_g
  for (int li = 0; li < 2; ++li)
    for (int j = 0; j < 12; ++j) Lp[li][j] = P(12 + li * 12 + j);
  const float* nn_b = P(36); const float* nn_g = P(37);
  const float* pW = P(38);   const float* pb = P(39);   const float* pv = P(40);
  const float* rW = P(41);   const float* rb = P(42);   const float* ru = P(43);

  // workspace carve (256B aligned)
  char* wsb = (char*)d_ws;
  size_t off = 0;
  auto carve = [&](size_t bytes) -> void* {
    void* p = wsb + off;
    off += (bytes + 255) & ~(size_t)255;
    return p;
  };
  float* Gf    = (float*)carve((size_t)Mrows * Dm * 4);
  h16*   Gh    = (h16*)  carve((size_t)Mrows * Dm * 2);
  h16*   QKVh  = (h16*)  carve((size_t)Mrows * 384 * 2);
  h16*   Ohh   = (h16*)  carve((size_t)Mrows * Dm * 2);
  h16*   FF1h  = (h16*)  carve((size_t)Mrows * FFd * 2);
  float* tmpF  = (float*)carve((size_t)Mrows * Dm * 4);
  h16* g0W1h = (h16*)carve(128 * 128 * 2);
  h16* g0W2h = (h16*)carve(128 * 128 * 2);
  h16* g1W1h = (h16*)carve(128 * 128 * 2);
  h16* g1W2h = (h16*)carve(128 * 128 * 2);
  h16* rWh   = (h16*)carve(128 * 128 * 2);
  h16* pWh   = (h16*)carve(128 * 128 * 2);
  h16* WqkvH[2]; h16* WoH[2]; h16* ffW1H[2]; h16* ffW2H[2];
  for (int li = 0; li < 2; ++li) {
    WqkvH[li] = (h16*)carve(384 * 128 * 2);
    WoH[li]   = (h16*)carve(128 * 128 * 2);
    ffW1H[li] = (h16*)carve(256 * 128 * 2);
    ffW2H[li] = (h16*)carve(128 * 256 * 2);
  }

  auto cvt = [&](const float* src, h16* dst, int N, int K, int Ksrc) {
    int tot = N * K;
    cvt_w<<<(tot + 255) / 256, 256, 0, stream>>>(src, dst, N, K, Ksrc);
  };
  cvt(g0W1, g0W1h, 128, 128, 3);     // zero-padded K: 3 -> 128
  cvt(g0W2, g0W2h, 128, 128, 128);
  cvt(g1W1, g1W1h, 128, 128, 128);
  cvt(g1W2, g1W2h, 128, 128, 128);
  cvt(rW, rWh, 128, 128, 128);
  cvt(pW, pWh, 128, 128, 128);
  for (int li = 0; li < 2; ++li) {
    cvt(Lp[li][3], WqkvH[li], 384, 128, 128);
    cvt(Lp[li][2], WoH[li],   128, 128, 128);
    cvt(Lp[li][0], ffW1H[li], 256, 128, 128);
    cvt(Lp[li][1], ffW2H[li], 128, 256, 256);
  }

  // stage 1: fused graph front-end -> G (with positional encoding)
  graph_kernel<<<Mrows, 256, GRAPH_LDS, stream>>>(
      x_enc, nn_g, nn_b,
      g0W1h, g0W2h, g0b1, g0b2, g0eps,
      g1W1h, g1W2h, g1b1, g1b2, g1eps,
      rWh, rb, ru, Gf, Gh);

  // stage 2: transformer encoder x2
  for (int li = 0; li < 2; ++li) {
    gemm_xwt<<<dim3(Mrows / 32, 384 / 64), 256, 0, stream>>>(
        Gh, WqkvH[li], Lp[li][7], nullptr, QKVh, 128, 384, 0);
    attn_kernel<<<dim3(16, HEADS, Bz), 32, ATT_LDS, stream>>>(QKVh, Ohh);
    gemm_xwt<<<dim3(Mrows / 32, 128 / 64), 256, 0, stream>>>(
        Ohh, WoH[li], Lp[li][6], tmpF, nullptr, 128, 128, 0);
    resid_ln<<<Mrows, Dm, 0, stream>>>(Gf, tmpF, Lp[li][9], Lp[li][8], Gf, Gh);
    gemm_xwt<<<dim3(Mrows / 32, 256 / 64), 256, 0, stream>>>(
        Gh, ffW1H[li], Lp[li][4], nullptr, FF1h, 128, 256, 1);      // gelu
    gemm_xwt<<<dim3(Mrows / 32, 128 / 64), 256, 0, stream>>>(
        FF1h, ffW2H[li], Lp[li][5], tmpF, nullptr, 256, 128, 0);
    resid_ln<<<Mrows, Dm, 0, stream>>>(Gf, tmpF, Lp[li][11], Lp[li][10], Gf, Gh);
  }

  // stage 3: attention pooling + classifier
  gemm_xwt<<<dim3(Mrows / 32, 128 / 64), 256, 0, stream>>>(
      Gh, pWh, pb, tmpF, nullptr, 128, 128, 2);                     // tanh
  pool_kernel<<<Bz, 256, 0, stream>>>(tmpF, Gf, pv, cls_W, cls_b, (float*)d_out);
}
